// SMOTEGCN_82497731822015
// MI455X (gfx1250) — matrix operand order repaired and checked by reading, not compiled, not run on hardware
//
#include <hip/hip_runtime.h>
#include <hip/hip_bf16.h>

// ---------------------------------------------------------------------------
// 2-layer GCN forward for MI455X (gfx1250, wave32).
//  GEMM1 (100000x512 @ 512x128) runs on v_wmma_f32_16x16x32_bf16 with fp32
//  accumulation; scatter phases use global_atomic_add_f32 (L2-resident).
// ---------------------------------------------------------------------------

#define EMB 512
#define HID 128
#define NOUT 3

typedef __bf16        v16bf __attribute__((ext_vector_type(16)));
typedef float         v8f   __attribute__((ext_vector_type(8)));
typedef unsigned int  u32x4 __attribute__((ext_vector_type(4)));
typedef float         f32x4 __attribute__((ext_vector_type(4)));

union Frag { u32x4 u[2]; v16bf v; };

__device__ __forceinline__ unsigned short f2bf_rne(float f) {
    unsigned int u = __float_as_uint(f);
    unsigned int r = u + 0x7fffu + ((u >> 16) & 1u);   // round-to-nearest-even
    return (unsigned short)(r >> 16);
}

// ---------------- init / normalization helpers ----------------

__global__ void k_fill_f32(float* __restrict__ p, size_t n, float v) {
    size_t i  = (size_t)blockIdx.x * blockDim.x + threadIdx.x;
    size_t st = (size_t)gridDim.x * blockDim.x;
    for (; i < n; i += st) p[i] = v;
}

__global__ void k_out_init(float* __restrict__ out, const float* __restrict__ b2, int n) {
    int i = blockIdx.x * blockDim.x + threadIdx.x;           // n = N*NOUT
    if (i < n) out[i] = b2[i % NOUT];
}

__global__ void k_deg(const int* __restrict__ dst, float* __restrict__ deg, int E) {
    int e = blockIdx.x * blockDim.x + threadIdx.x;
    if (e < E) atomicAdd(&deg[dst[e]], 1.0f);
}

__global__ void k_dinv(const float* __restrict__ deg, float* __restrict__ dinv, int N) {
    int i = blockIdx.x * blockDim.x + threadIdx.x;
    if (i < N) { float d = deg[i]; dinv[i] = d > 0.0f ? __frsqrt_rn(d) : 0.0f; }
}

__global__ void k_norm(const int* __restrict__ src, const int* __restrict__ dst,
                       const float* __restrict__ dinv, float* __restrict__ norm, int E) {
    int e = blockIdx.x * blockDim.x + threadIdx.x;
    if (e < E) norm[e] = dinv[src[e]] * dinv[dst[e]];
}

// ---------------- bf16 conversion / W1 packing ----------------

__global__ void k_cvt_bf16(const float* __restrict__ x, unsigned short* __restrict__ y, size_t n) {
    size_t i  = (size_t)blockIdx.x * blockDim.x + threadIdx.x;
    size_t st = (size_t)gridDim.x * blockDim.x;
    for (; i < n; i += st) y[i] = f2bf_rne(x[i]);
}

// Pack W1 (EMB x HID, row major) into this GEMM's B fragment order (contiguous k, same as A):
// chunk index = ((kt*8 + nt)*32 + lane)*16 ; element j of a chunk is
// W1[kt*32 + (lane<16?0:16) + j][nt*16 + (lane&15)], i.e. lane = N,
// VGPR pairs pack K (K=0..15 lanes 0-15, K=16..31 lanes 16-31).
__global__ void k_pack_w1(const float* __restrict__ W1, unsigned short* __restrict__ W1p) {
    int t = blockIdx.x * blockDim.x + threadIdx.x;           // 16*8*32*16 = 65536
    int j    =  t        & 15;
    int lane = (t >> 4)  & 31;
    int nt   = (t >> 9)  & 7;
    int kt   =  t >> 12;
    int k = kt * 32 + ((lane >> 4) << 4) + j;
    int n = nt * 16 + (lane & 15);
    W1p[t] = f2bf_rne(W1[(size_t)k * HID + n]);
}

// ---------------- GEMM1: h1 = Xbf @ W1p  (WMMA bf16 -> f32) ----------------

__global__ __launch_bounds__(256, 2)
void k_gemm1(const unsigned short* __restrict__ Xbf,
             const unsigned short* __restrict__ W1p,
             float* __restrict__ h1) {
    const int mt    = blockIdx.x;              // 16-row slab
    const int wave  = threadIdx.x >> 5;        // N-tile 0..7  (HID = 8*16)
    const int lane  = threadIdx.x & 31;
    const int m     = lane & 15;
    const int khalf = (lane >> 4) << 4;        // 0 or 16 (A-layout lane split)

    const unsigned short* arow = Xbf + ((size_t)mt * 16 + m) * EMB;

    v8f c = {};
#pragma unroll 4
    for (int kt = 0; kt < EMB / 32; ++kt) {
        Frag a, b;
        const unsigned short* ap = arow + kt * 32 + khalf;
        a.u[0] = *(const u32x4*)(ap);          // K = khalf+0..7
        a.u[1] = *(const u32x4*)(ap + 8);      // K = khalf+8..15
        __builtin_prefetch(ap + 32, 0, 1);     // next K step -> global_prefetch_b8

        const unsigned short* bp = W1p + (((size_t)kt * 8 + wave) * 32 + lane) * 16;
        b.u[0] = *(const u32x4*)(bp);          // K = base+0..7  in v0..v3
        b.u[1] = *(const u32x4*)(bp + 8);      // K = base+8..15 in v4..v7

        c = __builtin_amdgcn_wmma_f32_16x16x32_bf16(
                false, a.v, false, b.v, (short)0, c, false, false);
    }

    const int nbase = wave * 16 + (lane & 15);
    const int rbase = mt * 16 + ((lane >> 4) << 3);   // lanes 16-31 hold M=8..15
#pragma unroll
    for (int r = 0; r < 8; ++r)
        h1[(size_t)(rbase + r) * HID + nbase] = c[r];
}

// ---------------- scatter-add layer 1 (wave per edge, 4 floats/lane) -------

__global__ void k_scatter1(const float* __restrict__ h1, const float* __restrict__ norm,
                           const float* __restrict__ dinv,
                           const int* __restrict__ src, const int* __restrict__ dst,
                           float* __restrict__ agg, int E, int T) {
    int g    = (blockIdx.x * blockDim.x + threadIdx.x) >> 5;
    int lane = threadIdx.x & 31;
    if (g >= T) return;
    int s, d; float w;
    if (g < E) { s = src[g]; d = dst[g]; w = norm[g]; }
    else       { s = d = g - E; float di = dinv[s]; w = di * di; }   // self loop
    f32x4 v = *(const f32x4*)(h1 + (size_t)s * HID + lane * 4);
    float* o = agg + (size_t)d * HID + lane * 4;
    atomicAdd(o + 0, v.x * w);
    atomicAdd(o + 1, v.y * w);
    atomicAdd(o + 2, v.z * w);
    atomicAdd(o + 3, v.w * w);
}

__global__ void k_relu_bias(float* __restrict__ agg, const float* __restrict__ b1, size_t n) {
    size_t i  = (size_t)blockIdx.x * blockDim.x + threadIdx.x;
    size_t st = (size_t)gridDim.x * blockDim.x;
    for (; i < n; i += st) {
        float x = agg[i] + b1[i & (HID - 1)];
        agg[i] = x > 0.0f ? x : 0.0f;
    }
}

// ---------------- GEMM2: h2 = x1 @ W2  (N=3, LDS tree reduce) --------------

__global__ __launch_bounds__(HID)
void k_h2(const float* __restrict__ x1, const float* __restrict__ W2,
          float* __restrict__ h2) {
    __shared__ float s[NOUT][HID];
    int n = blockIdx.x, t = threadIdx.x;
    float v = x1[(size_t)n * HID + t];
    s[0][t] = v * W2[t * NOUT + 0];
    s[1][t] = v * W2[t * NOUT + 1];
    s[2][t] = v * W2[t * NOUT + 2];
    __syncthreads();
    for (int off = HID / 2; off > 0; off >>= 1) {
        if (t < off) {
            s[0][t] += s[0][t + off];
            s[1][t] += s[1][t + off];
            s[2][t] += s[2][t + off];
        }
        __syncthreads();
    }
    if (t == 0) {
        h2[(size_t)n * NOUT + 0] = s[0][0];
        h2[(size_t)n * NOUT + 1] = s[1][0];
        h2[(size_t)n * NOUT + 2] = s[2][0];
    }
}

__global__ void k_scatter2(const float* __restrict__ h2, const float* __restrict__ norm,
                           const float* __restrict__ dinv,
                           const int* __restrict__ src, const int* __restrict__ dst,
                           float* __restrict__ out, int E, int T) {
    int g = blockIdx.x * blockDim.x + threadIdx.x;
    if (g >= T) return;
    int s, d; float w;
    if (g < E) { s = src[g]; d = dst[g]; w = norm[g]; }
    else       { s = d = g - E; float di = dinv[s]; w = di * di; }
#pragma unroll
    for (int j = 0; j < NOUT; ++j)
        atomicAdd(&out[(size_t)d * NOUT + j], w * h2[(size_t)s * NOUT + j]);
}

// ---------------------------------------------------------------------------

extern "C" void kernel_launch(void* const* d_in, const int* in_sizes, int n_in,
                              void* d_out, int out_size, void* d_ws, size_t ws_size,
                              hipStream_t stream) {
    const float* feature = (const float*)d_in[0];
    const float* W1      = (const float*)d_in[1];
    const float* b1      = (const float*)d_in[2];
    const float* W2      = (const float*)d_in[3];
    const float* b2      = (const float*)d_in[4];
    const int*   eidx    = (const int*)d_in[5];
    const int    E       = in_sizes[6];                  // edge_type length
    const int    N       = in_sizes[0] / EMB;            // 100000
    const int*   src     = eidx;
    const int*   dst     = eidx + E;
    float*       out     = (float*)d_out;

    // -------- workspace carve-out (256B aligned) --------
    char*  ws  = (char*)d_ws;
    size_t off = 0;
    auto carve = [&](size_t bytes) { size_t o = off; off = (off + bytes + 255) & ~(size_t)255; return o; };
    unsigned short* Xbf  = (unsigned short*)(ws + carve((size_t)N * EMB * 2));
    float*          agg1 = (float*)Xbf;                  // alias: Xbf dead after GEMM1
    float*          h1   = (float*)(ws + carve((size_t)N * HID * 4));
    float*          deg  = (float*)(ws + carve((size_t)N * 4));
    float*          dinv = (float*)(ws + carve((size_t)N * 4));
    float*          norm = (float*)(ws + carve((size_t)E * 4));
    float*          h2   = (float*)(ws + carve((size_t)N * NOUT * 4));
    unsigned short* W1p  = (unsigned short*)(ws + carve((size_t)EMB * HID * 2));
    (void)ws_size; (void)n_in; (void)out_size;

    const int T = E + N;                                  // edges + self loops

    // degrees / symmetric norm
    k_fill_f32<<<1024, 256, 0, stream>>>(deg, (size_t)N, 1.0f);   // self loop
    k_deg <<<(E + 255) / 256, 256, 0, stream>>>(dst, deg, E);
    k_dinv<<<(N + 255) / 256, 256, 0, stream>>>(deg, dinv, N);
    k_norm<<<(E + 255) / 256, 256, 0, stream>>>(src, dst, dinv, norm, E);

    // bf16 conversion + B-layout pack
    k_cvt_bf16<<<4096, 256, 0, stream>>>(feature, Xbf, (size_t)N * EMB);
    k_pack_w1 <<<(EMB / 32) * 8 * 32 * 16 / 256, 256, 0, stream>>>(W1, W1p);

    // layer-1 GEMM on WMMA
    k_gemm1<<<N / 16, 256, 0, stream>>>(Xbf, W1p, h1);

    // layer-1 scatter (agg1 aliases Xbf; zero after GEMM, before scatter)
    k_fill_f32<<<4096, 256, 0, stream>>>(agg1, (size_t)N * HID, 0.0f);
    {
        long long thr = (long long)T * 32;
        k_scatter1<<<(int)((thr + 255) / 256), 256, 0, stream>>>(h1, norm, dinv, src, dst, agg1, E, T);
    }
    k_relu_bias<<<4096, 256, 0, stream>>>(agg1, b1, (size_t)N * HID);

    // layer-2 transform + scatter (+b2 folded into out init)
    k_h2<<<N, HID, 0, stream>>>(agg1, W2, h2);
    k_out_init<<<(N * NOUT + 255) / 256, 256, 0, stream>>>(out, b2, N * NOUT);
    k_scatter2<<<(T + 255) / 256, 256, 0, stream>>>(h2, norm, dinv, src, dst, out, E, T);
}
